// TopKSAE_50002009260540
// MI455X (gfx1250) — matrix operand order repaired and checked by reading, not compiled, not run on hardware
//
#include <hip/hip_runtime.h>
#include <hip/hip_bf16.h>

// ---------------------------------------------------------------------------
// TopK SAE forward for MI455X (gfx1250, wave32, WMMA).
//   recon [8192,768], acts [8192,24576]   (d_out = recon ++ acts, fp32)
// Pipeline:
//   k0: fp32 -> bf16 (RNE) for x and W_enc               (~100 MB traffic)
//   k1: encoder GEMM via v_wmma_f32_16x16x32_bf16, +bias -> pre_acts
//       written straight into the acts region of d_out   (805 MB write)
//       -- k-loop fully unrolled (IN_DIM/32 = 24 steps of straight-line
//          code): no backedge register constraints -> scheduler pipelines
//          load clauses ahead of WMMAs with partial loadcnt waits, no
//          rotation copies, no hazard nops
//   k2: exact top-64 radix select per row in LDS; rewrite row as
//       relu/zero scatter; emit (idx,val) pairs to ws    (805 MB R + 805 MB W)
//   k3: sparse decode: recon = b_dec + sum_k val*W_dec[idx]  (L2-resident)
// ---------------------------------------------------------------------------

#define IN_DIM 768
#define DICT   24576
#define BATCH  8192
#define TK     64

typedef __attribute__((ext_vector_type(16))) __bf16          v16bf;
typedef __attribute__((ext_vector_type(8)))  float           v8f;
typedef __attribute__((ext_vector_type(8)))  unsigned short  u16x8;
typedef __attribute__((ext_vector_type(16))) unsigned short  u16x16;

// ---------------- k0: fp32 -> bf16 (round to nearest even) -----------------
__global__ __launch_bounds__(256) void f32_to_bf16_kernel(
    const float* __restrict__ in, unsigned short* __restrict__ out, int n) {
  int i = blockIdx.x * 256 + threadIdx.x;
  if (i < n) {
    unsigned int u = __builtin_bit_cast(unsigned int, in[i]);
    unsigned int r = (u + 0x7FFFu + ((u >> 16) & 1u)) >> 16;   // RNE
    out[i] = (unsigned short)r;
  }
}

// A-operand fragment: row = lane%16, K chunks {base..base+7},{base+16..+23}
__device__ __forceinline__ u16x16 load_a_frag(const unsigned short* pA) {
  u16x8 lo = *(const u16x8*)(pA);
  u16x8 hi = *(const u16x8*)(pA + 16);
  return __builtin_shufflevector(lo, hi,
      0, 1, 2, 3, 4, 5, 6, 7, 8, 9, 10, 11, 12, 13, 14, 15);
}

__device__ __forceinline__ v8f wmma_bf16(u16x16 a, u16x16 b, v8f c) {
  return __builtin_amdgcn_wmma_f32_16x16x32_bf16(
      /*neg_a=*/false, __builtin_bit_cast(v16bf, a),
      /*neg_b=*/false, __builtin_bit_cast(v16bf, b),
      /*c_mod=*/(short)0, c, /*reuse_a=*/false, /*reuse_b=*/false);
}

// ---------------- k1: encoder GEMM (NT, bf16 WMMA, f32 acc) ----------------
// Each wave: one 16-col dict tile x four 16-row batch tiles (B reused 4x
// from registers; W_enc/x are L2-resident). Block = 8 waves = 128 dict cols.
__global__ __launch_bounds__(256) void sae_enc_wmma(
    const unsigned short* __restrict__ Xb,   // [BATCH, IN_DIM] bf16
    const unsigned short* __restrict__ Wb,   // [DICT,  IN_DIM] bf16
    const float* __restrict__ bias,          // [DICT]
    float* __restrict__ pre)                 // [BATCH, DICT] (acts region)
{
  const int wave = threadIdx.x >> 5;
  const int lane = threadIdx.x & 31;
  const int half = lane >> 4;      // lane group 0/1
  const int l16  = lane & 15;
  const int n0 = (blockIdx.x * 8 + wave) * 16;
  const int m0 = blockIdx.y * 64;

  // B operand: col = l16, K chunks {base..base+7},{base+16..+23} (matches A)
  const unsigned short* pB  = Wb + (size_t)(n0 + l16) * IN_DIM + half * 8;
  // A operand base: row = l16
  const unsigned short* pA0 = Xb + (size_t)(m0 + l16) * IN_DIM + half * 8;

  v8f acc[4] = {};

  // Fully unrolled straight-line k-loop: 24 steps x (10 b128 loads + 4 WMMA).
  // The scheduler hoists load clauses ahead and emits partial loadcnt waits.
#pragma unroll
  for (int k = 0; k < IN_DIM; k += 32) {
    u16x16 b = load_a_frag(pB + k);
    u16x16 a[4];
#pragma unroll
    for (int s = 0; s < 4; ++s)
      a[s] = load_a_frag(pA0 + (size_t)(s * 16) * IN_DIM + k);
#pragma unroll
    for (int s = 0; s < 4; ++s)
      acc[s] = wmma_bf16(a[s], b, acc[s]);
  }

  const float bn = bias[n0 + l16];
#pragma unroll
  for (int s = 0; s < 4; ++s) {
#pragma unroll
    for (int i = 0; i < 8; ++i) {
      const int m = m0 + s * 16 + i + 8 * half;           // C layout
      pre[(size_t)m * DICT + (n0 + l16)] = acc[s][i] + bn;
    }
  }
}

// ---------------- k2: exact top-64 per row (LDS radix select) --------------
__global__ __launch_bounds__(256) void sae_topk(
    float* __restrict__ acts,                // in: pre_acts, out: sparse acts
    int* __restrict__ tk_idx, float* __restrict__ tk_val)
{
  extern __shared__ char smem[];
  float* row = (float*)smem;                                   // 24576 f32
  unsigned int* hist = (unsigned int*)(smem + DICT * sizeof(float)); // 256
  __shared__ unsigned int s_prefix, s_want, s_eqTaken, s_selCount;

  const int b = blockIdx.x;
  float* grow = acts + (size_t)b * DICT;

  for (int i = threadIdx.x; i < DICT; i += 256) row[i] = grow[i];
  if (threadIdx.x == 0) { s_prefix = 0; s_want = TK; s_eqTaken = 0; s_selCount = 0; }
  __syncthreads();

  // 4-round radix select on sortable keys (descending).
  for (int round = 0; round < 4; ++round) {
    const int shift = 24 - round * 8;
    for (int i = threadIdx.x; i < 256; i += 256) hist[i] = 0;
    __syncthreads();
    const unsigned int prefix = s_prefix;
    for (int i = threadIdx.x; i < DICT; i += 256) {
      unsigned int u = __builtin_bit_cast(unsigned int, row[i]);
      unsigned int key = u ^ (((int)u < 0) ? 0xFFFFFFFFu : 0x80000000u);
      if (round == 0 || (key >> (shift + 8)) == prefix)
        atomicAdd(&hist[(key >> shift) & 255u], 1u);
    }
    __syncthreads();
    if (threadIdx.x == 0) {
      unsigned int want = s_want, cum = 0; int sel = 0;
      for (int bin = 255; bin >= 0; --bin) {
        if (cum + hist[bin] >= want) { sel = bin; break; }
        cum += hist[bin];
      }
      s_prefix = (prefix << 8) | (unsigned int)sel;
      s_want = want - cum;                 // how many to take at == threshold
    }
    __syncthreads();
  }

  const unsigned int T = s_prefix;         // key of 64th-largest element
  const unsigned int wantEq = s_want;
  for (int i = threadIdx.x; i < DICT; i += 256) {
    float v = row[i];
    unsigned int u = __builtin_bit_cast(unsigned int, v);
    unsigned int key = u ^ (((int)u < 0) ? 0xFFFFFFFFu : 0x80000000u);
    bool sel = false;
    if (key > T)       sel = true;
    else if (key == T) sel = (atomicAdd(&s_eqTaken, 1u) < wantEq);
    float outv = sel ? fmaxf(v, 0.0f) : 0.0f;        // ReLU on selected only
    grow[i] = outv;
    if (sel) {
      unsigned int p = atomicAdd(&s_selCount, 1u);
      if (p < TK) { tk_idx[b * TK + p] = i; tk_val[b * TK + p] = outv; }
    }
  }
  __syncthreads();
  for (unsigned int p = s_selCount + threadIdx.x; p < TK; p += 256) {
    tk_idx[b * TK + p] = 0; tk_val[b * TK + p] = 0.0f;   // safety padding
  }
}

// ---------------- k3: sparse decode (W_dec gathers hit L2) -----------------
__global__ __launch_bounds__(256) void sae_decode(
    const int* __restrict__ tk_idx, const float* __restrict__ tk_val,
    const float* __restrict__ Wdec, const float* __restrict__ bdec,
    float* __restrict__ recon)
{
  __shared__ int   sidx[TK];
  __shared__ float sval[TK];
  const int b = blockIdx.x;
  if (threadIdx.x < TK) {
    sidx[threadIdx.x] = tk_idx[b * TK + threadIdx.x];
    sval[threadIdx.x] = tk_val[b * TK + threadIdx.x];
  }
  __syncthreads();
  const int c0 = threadIdx.x, c1 = threadIdx.x + 256, c2 = threadIdx.x + 512;
  float a0 = bdec[c0], a1 = bdec[c1], a2 = bdec[c2];
  for (int t = 0; t < TK; ++t) {
    const float* w = Wdec + (size_t)sidx[t] * IN_DIM;
    const float v = sval[t];
    a0 = fmaf(v, w[c0], a0);
    a1 = fmaf(v, w[c1], a1);
    a2 = fmaf(v, w[c2], a2);
  }
  float* r = recon + (size_t)b * IN_DIM;
  r[c0] = a0; r[c1] = a1; r[c2] = a2;
}

// ---------------------------------------------------------------------------
extern "C" void kernel_launch(void* const* d_in, const int* in_sizes, int n_in,
                              void* d_out, int out_size, void* d_ws, size_t ws_size,
                              hipStream_t stream) {
  const float* x    = (const float*)d_in[0];   // [8192,768]
  const float* Wenc = (const float*)d_in[1];   // [24576,768]
  const float* benc = (const float*)d_in[2];   // [24576]
  const float* Wdec = (const float*)d_in[3];   // [24576,768]
  const float* bdec = (const float*)d_in[4];   // [768]

  float* recon = (float*)d_out;                       // [8192,768]
  float* acts  = recon + (size_t)BATCH * IN_DIM;      // [8192,24576]

  // Workspace: bf16 copies of x and W_enc, plus top-k (idx,val) pairs. ~54 MB.
  unsigned short* Xb = (unsigned short*)d_ws;
  unsigned short* Wb = Xb + (size_t)BATCH * IN_DIM;
  int*   tk_idx = (int*)  (Wb + (size_t)DICT * IN_DIM);
  float* tk_val = (float*)(tk_idx + (size_t)BATCH * TK);

  {
    int n = BATCH * IN_DIM;
    f32_to_bf16_kernel<<<(n + 255) / 256, 256, 0, stream>>>(x, Xb, n);
  }
  {
    int n = DICT * IN_DIM;
    f32_to_bf16_kernel<<<(n + 255) / 256, 256, 0, stream>>>(Wenc, Wb, n);
  }

  dim3 gEnc(DICT / 128, BATCH / 64);   // (192, 128), 8 waves/block
  sae_enc_wmma<<<gEnc, 256, 0, stream>>>(Xb, Wb, benc, acts);

  const size_t smem = (size_t)DICT * sizeof(float) + 256 * sizeof(unsigned int);
  sae_topk<<<BATCH, 256, smem, stream>>>(acts, tk_idx, tk_val);

  sae_decode<<<BATCH, 256, 0, stream>>>(tk_idx, tk_val, Wdec, bdec, recon);
}
